// LTIModel_20933670600776
// MI455X (gfx1250) — compile-verified, matrix operand order checked by reading, not run on hardware
//
#include <hip/hip_runtime.h>
#include <hip/hip_bf16.h>
#include <math.h>

// ---------------------------------------------------------------------------
// LTI controller scan + value MLP for MI455X (gfx1250, wave32, WMMA, TDM).
//
//  * RK4 on a linear system collapses to x_{t+1} = x Phi + y Gam with
//    Phi = I + hM + h^2M^2/2 + h^3M^3/6 + h^4M^4/24 (M = A_T), precomputed
//    once in fp32 on device (Horner), then packed to f16 WMMA-B layout.
//  * Scan is latency bound: 2 persistent workgroups (16 batch rows each),
//    Phi/Gam/Cu/Duy resident in 240KB of the WGP's 320KB LDS, ping-pong
//    f16 x/y buffers, one barrier per step, 12 v_wmma per tile per step.
//  * Weight upload into LDS additionally issued through the Tensor Data
//    Mover (tensor_load_to_lds + s_wait_tensorcnt); a plain copy loop writes
//    the same bytes so the kernel is correct even if the (arity-only-probed)
//    descriptor interpretation is off.
//  * Value MLP: separate WMMA kernel, 16-row tile per wave, tanh in VALU,
//    final dot via wave32 shfl_xor reduction.
// ---------------------------------------------------------------------------

typedef __attribute__((ext_vector_type(16))) _Float16 v16h;
typedef __attribute__((ext_vector_type(8)))  float    v8f;
typedef __attribute__((ext_vector_type(4)))  unsigned int u32x4;
typedef __attribute__((ext_vector_type(8)))  int      i32x8;
typedef __attribute__((ext_vector_type(4)))  int      i32x4;

#define Bsz   32
#define Tsz   2048
#define OBSsz 128
#define Nsz   256
#define OUTsz 32
#define Hsz   64

__device__ __forceinline__ v8f wmma16(v16h a, v16h b, v8f c) {
  return __builtin_amdgcn_wmma_f32_16x16x32_f16(
      /*neg_a=*/false, a, /*neg_b=*/false, b,
      /*c_mod=*/(short)0, c, /*reuse_a=*/false, /*reuse_b=*/false);
}

// A-fragment (16x32, MxK) from row-major f16 LDS. ISA layout:
// lanes 0-15 (m=lane): VGPR0..3 hold K 0..7, VGPR4..7 hold K 16..23
// lanes 16-31 (m=lane-16): K 8..15 and 24..31.  Pairs are contiguous -> b32 loads.
__device__ __forceinline__ v16h load_a(const _Float16* base, int ldk, int kbase, int lane) {
  int m = lane & 15;
  int hb = (lane >> 4) << 3;
  const _Float16* row = base + m * ldk + kbase + hb;
  union { v16h v; unsigned u[8]; } a;
#pragma unroll
  for (int vv = 0; vv < 8; ++vv) {
    int e = vv << 1;
    int k = (e & 7) + ((e >> 3) << 4);
    a.u[vv] = *(const unsigned*)(row + k);
  }
  return a.v;
}

// B-fragment (32x16, KxN) from pre-swizzled pack: per 32x16 tile, 32 lanes x
// 16 contiguous f16 (32B per lane) -> two ds_load_b128 per lane.
__device__ __forceinline__ v16h load_b(const _Float16* pack, int tileCols, int kt, int nt, int lane) {
  const _Float16* p = pack + (((kt * tileCols + nt) << 9) + (lane << 4));
  union { v16h v; uint4 q[2]; } b;
  b.q[0] = ((const uint4*)p)[0];
  b.q[1] = ((const uint4*)p)[1];
  return b.v;
}

// ------------------------- one-time precompute kernels ---------------------

// C[i][j] = (addI && i==j) + alpha * sum_k A[i][k]*Bm[k][j]
__global__ void mm_f32(const float* __restrict__ A, const float* __restrict__ Bm,
                       float* __restrict__ C, int M, int K, int N, float alpha, int addI) {
  int idx = blockIdx.x * blockDim.x + threadIdx.x;
  if (idx >= M * N) return;
  int i = idx / N, j = idx % N;
  float s = 0.f;
  for (int k = 0; k < K; ++k) s += A[i * K + k] * Bm[k * N + j];
  C[idx] = alpha * s + ((addI && i == j) ? 1.0f : 0.0f);
}

// C = I + c*M   (square NxN)
__global__ void eltI_f32(const float* __restrict__ M, float* __restrict__ C, float c, int N) {
  int idx = blockIdx.x * blockDim.x + threadIdx.x;
  if (idx >= N * N) return;
  int i = idx / N, j = idx % N;
  C[idx] = c * M[idx] + ((i == j) ? 1.0f : 0.0f);
}

// Swizzle row-major f32 [K][N] into f16 WMMA-B fragment-major pack.
__global__ void pack_b_f16(const float* __restrict__ src, _Float16* __restrict__ dst, int K, int N) {
  int idx = blockIdx.x * blockDim.x + threadIdx.x;
  if (idx >= K * N) return;
  int tileCols = N >> 4;
  int tileId = idx >> 9;
  int lane = (idx >> 4) & 31;
  int e = idx & 15;
  int kt = tileId / tileCols, nt = tileId % tileCols;
  int k = kt * 32 + (e & 7) + ((e >> 3) << 4) + ((lane >> 4) << 3);
  int n = nt * 16 + (lane & 15);
  dst[idx] = (_Float16)src[k * N + n];
}

// ------------------------------ the scan -----------------------------------
// grid = 2 workgroups (16 batch rows each), block = 512 (16 waves).
// LDS: PhiP 128KB | GamP 64KB | CuP 16KB | DuyP 8KB | x ping-pong 16KB | y ping-pong 8KB = 240KB.
__launch_bounds__(512, 1)
__global__ void scan_kernel(const float* __restrict__ obs, const float* __restrict__ x0,
                            float* __restrict__ outp,
                            const _Float16* __restrict__ packW /* Phi|Gam|Cu|Duy contiguous */) {
  extern __shared__ char smem[];
  _Float16* sPhi = (_Float16*)smem;       // 65536 halves
  _Float16* sGam = sPhi + 65536;          // 32768
  _Float16* sCu  = sGam + 32768;          // 8192
  _Float16* sDuy = sCu  + 8192;           // 4096
  _Float16* sX   = sDuy + 4096;           // 2 * 16*256
  _Float16* sY   = sX   + 8192;           // 2 * 16*128

  const int tid = threadIdx.x;
  const int lane = tid & 31;
  const int w = tid >> 5;
  const int b0r = blockIdx.x * 16;

  // ---- weight upload: TDM descriptor per ISA 8.3/8.4 (221184B = 1024x27 of 8B)
  if (w == 0) {
    unsigned lds_off = (unsigned)(size_t)sPhi;              // LDS aperture addr[31:0]
    unsigned long long ga = (unsigned long long)(size_t)packW;
    u32x4 g0;
    g0[0] = 1u;                                             // count=1, is_restore=0
    g0[1] = lds_off;                                        // lds_addr (bytes)
    g0[2] = (unsigned)(ga & 0xFFFFFFFFu);                   // global_addr[31:0]
    g0[3] = (unsigned)((ga >> 32) & 0x01FFFFFFu) | (2u << 30);  // global_addr[56:32] | type=2
    const unsigned td0 = 1024u, td1 = 27u, tile0 = 1024u, tile1 = 27u;
    const unsigned long long st0 = 1024ull;                 // dim0 stride (8B units)
    i32x8 g1;
    g1[0] = (int)(3u << 16);                                // wg_mask=0, data_size=8B, no flags
    g1[1] = (int)((td0 & 0xFFFFu) << 16);                   // barrier_addr=0 | tensor_dim0[15:0]
    g1[2] = (int)((td0 >> 16) | ((td1 & 0xFFFFu) << 16));   // tensor_dim0[31:16] | tensor_dim1[15:0]
    g1[3] = (int)((td1 >> 16) | (tile0 << 16));             // tensor_dim1[31:16] | tile_dim0
    g1[4] = (int)(tile1);                                   // tile_dim1 | tile_dim2=0
    g1[5] = (int)(st0 & 0xFFFFFFFFu);                       // stride0[31:0]
    g1[6] = (int)(st0 >> 32);                               // stride0[47:32] | stride1[15:0]=0
    g1[7] = 0;                                              // stride1[47:16]=0
    i32x4 gz = {0, 0, 0, 0};
    i32x8 gz8 = {0, 0, 0, 0, 0, 0, 0, 0};
    // this toolchain declares the 6-arg (clang-23 / therock) form
    __builtin_amdgcn_tensor_load_to_lds(g0, g1, gz, gz, gz8, 0);
  }
  // Belt-and-braces copy of the same bytes (13824 uint4); TDM writes identical
  // data so the overlap is benign, and this is one-time work.
  {
    const uint4* s = (const uint4*)packW;
    uint4* d = (uint4*)sPhi;
    for (int i = tid; i < 13824; i += 512) d[i] = s[i];
  }
  // x0 -> sX[buf0] (f16)
  for (int i = tid; i < 16 * Nsz; i += 512) {
    int r = i >> 8, c = i & 255;
    sX[i] = (_Float16)x0[(b0r + r) * Nsz + c];
  }
  // y0 -> sY[buf0]
  for (int i = tid; i < 16 * OBSsz; i += 512) {
    int r = i >> 7, c = i & 127;
    sY[i] = (_Float16)obs[((size_t)(b0r + r) * Tsz) * OBSsz + c];
  }
  if (w == 0) __builtin_amdgcn_s_wait_tensorcnt(0);
  __syncthreads();

  const int n = lane & 15;
  const int half = lane >> 4;

  for (int t = 0; t < Tsz; ++t) {
    const int cur = t & 1, nxt = cur ^ 1;
    const _Float16* xc = sX + cur * 4096;
    _Float16* xn = sX + nxt * 4096;
    const _Float16* yc = sY + cur * 2048;
    _Float16* yn = sY + nxt * 2048;

    // prefetch y_{t+1} into registers (wave w owns row w), hint t+2 into L2
    float4 yv = make_float4(0.f, 0.f, 0.f, 0.f);
    if (t + 1 < Tsz) {
      const float* p = obs + ((size_t)(b0r + w) * Tsz + (t + 1)) * OBSsz + lane * 4;
      yv = *(const float4*)p;
      if (t + 2 < Tsz) __builtin_prefetch(p + OBSsz, 0, 3);
    }

    // actions u_t = x_t@Cu + y_t@Duy : waves 0,1 own the two 16x16 tiles
    if (w < 2) {
      v8f acc = {};
#pragma unroll
      for (int kt = 0; kt < 8; ++kt)
        acc = wmma16(load_a(xc, Nsz, kt * 32, lane), load_b(sCu, 2, kt, w, lane), acc);
#pragma unroll
      for (int kt = 0; kt < 4; ++kt)
        acc = wmma16(load_a(yc, OBSsz, kt * 32, lane), load_b(sDuy, 2, kt, w, lane), acc);
#pragma unroll
      for (int r = 0; r < 8; ++r) {
        int m = half * 8 + r;
        outp[((size_t)(b0r + m) * Tsz + t) * 64 + w * 16 + n] = acc[r];
      }
    }

    // x_{t+1} tile (cols w*16..w*16+15) = x_t@Phi + y_t@Gam
    v8f xacc = {};
#pragma unroll
    for (int kt = 0; kt < 8; ++kt)
      xacc = wmma16(load_a(xc, Nsz, kt * 32, lane), load_b(sPhi, 16, kt, w, lane), xacc);
#pragma unroll
    for (int kt = 0; kt < 4; ++kt)
      xacc = wmma16(load_a(yc, OBSsz, kt * 32, lane), load_b(sGam, 16, kt, w, lane), xacc);

#pragma unroll
    for (int r = 0; r < 8; ++r) {
      int m = half * 8 + r;
      xn[m * Nsz + w * 16 + n] = (_Float16)xacc[r];
    }
    if (t == Tsz - 1) {  // x_final in f32 straight from accumulators
#pragma unroll
      for (int r = 0; r < 8; ++r) {
        int m = half * 8 + r;
        outp[(size_t)Bsz * Tsz * 64 + (size_t)(b0r + m) * Nsz + w * 16 + n] = xacc[r];
      }
    }
    // commit prefetched y_{t+1}
    if (t + 1 < Tsz) {
      _Float16* q = yn + w * OBSsz + lane * 4;
      q[0] = (_Float16)yv.x; q[1] = (_Float16)yv.y;
      q[2] = (_Float16)yv.z; q[3] = (_Float16)yv.w;
    }
    __syncthreads();
  }
}

// ------------------------------ log-std fill -------------------------------
__global__ void fill_logstd(const float* __restrict__ ls, float* __restrict__ outp) {
  int i = blockIdx.x * blockDim.x + threadIdx.x;
  if (i >= Bsz * Tsz * OUTsz) return;
  int bt = i >> 5, o = i & 31;
  outp[(size_t)bt * 64 + 32 + o] = ls[o];
}

// ------------------------------ value MLP ----------------------------------
// block = 512 (16 waves), wave -> 16-row tile, grid = 65536/256 = 256.
__launch_bounds__(512, 1)
__global__ void mlp_kernel(const float* __restrict__ obs,
                           const _Float16* __restrict__ W01P /* W0P|W1P contiguous */,
                           const float* __restrict__ b0v, const float* __restrict__ b1v,
                           const float* __restrict__ W2v, const float* __restrict__ b2v,
                           float* __restrict__ outv) {
  extern __shared__ char smem[];
  _Float16* sW0 = (_Float16*)smem;          // 8192 halves
  _Float16* sW1 = sW0 + 8192;               // 4096
  float* sB0 = (float*)(sW1 + 4096);        // 64
  float* sB1 = sB0 + 64;                    // 64
  float* sW2 = sB1 + 64;                    // 64
  _Float16* sTile = (_Float16*)(sW2 + 64);  // 16 waves * 16*128 halves

  const int tid = threadIdx.x;
  const int lane = tid & 31;
  const int w = tid >> 5;

  {
    const uint4* s = (const uint4*)W01P;
    uint4* d = (uint4*)sW0;
    for (int i = tid; i < (8192 + 4096) / 8; i += 512) d[i] = s[i];
  }
  for (int i = tid; i < 64; i += 512) { sB0[i] = b0v[i]; sB1[i] = b1v[i]; sW2[i] = W2v[i]; }
  __syncthreads();
  const float b2s = b2v[0];

  const int rowBase = blockIdx.x * 256 + w * 16;
  _Float16* myT = sTile + w * 2048;

  // stage obs tile [16][128] f32->f16 (coalesced 512B per wave per row)
  for (int r = 0; r < 16; ++r) {
    float4 v = *(const float4*)(obs + (size_t)(rowBase + r) * OBSsz + lane * 4);
    _Float16* q = myT + r * 128 + lane * 4;
    q[0] = (_Float16)v.x; q[1] = (_Float16)v.y; q[2] = (_Float16)v.z; q[3] = (_Float16)v.w;
  }

  const int n = lane & 15;
  const int half = lane >> 4;

  // layer 1: h1 = tanh(obs@W0 + b0), 4 col-tiles, K=128
  v8f h1[4];
#pragma unroll
  for (int nt = 0; nt < 4; ++nt) {
    v8f acc = {};
#pragma unroll
    for (int kt = 0; kt < 4; ++kt)
      acc = wmma16(load_a(myT, 128, kt * 32, lane), load_b(sW0, 4, kt, nt, lane), acc);
    h1[nt] = acc;
  }
  // tanh + f16, overwrite cols 0..63 of the staged tile (obs reads all done)
#pragma unroll
  for (int nt = 0; nt < 4; ++nt) {
    float bb = sB0[nt * 16 + n];
#pragma unroll
    for (int r = 0; r < 8; ++r) {
      int m = half * 8 + r;
      myT[m * 128 + nt * 16 + n] = (_Float16)tanhf(h1[nt][r] + bb);
    }
  }

  // layer 2: h2 = tanh(h1@W1 + b1), K=64, then dot with W2 via shuffle reduce
  float part[8];
#pragma unroll
  for (int r = 0; r < 8; ++r) part[r] = 0.f;
#pragma unroll
  for (int nt = 0; nt < 4; ++nt) {
    v8f acc = {};
#pragma unroll
    for (int kt = 0; kt < 2; ++kt)
      acc = wmma16(load_a(myT, 128, kt * 32, lane), load_b(sW1, 4, kt, nt, lane), acc);
    float bb = sB1[nt * 16 + n];
    float ww = sW2[nt * 16 + n];
#pragma unroll
    for (int r = 0; r < 8; ++r) part[r] += tanhf(acc[r] + bb) * ww;
  }
  // reduce over the 16 lanes of each half (wave32)
#pragma unroll
  for (int off = 1; off <= 8; off <<= 1)
#pragma unroll
    for (int r = 0; r < 8; ++r) part[r] += __shfl_xor(part[r], off, 32);

  if ((lane & 15) == 0) {
#pragma unroll
    for (int r = 0; r < 8; ++r) outv[rowBase + half * 8 + r] = part[r] + b2s;
  }
}

// ------------------------------ host side ----------------------------------
extern "C" void kernel_launch(void* const* d_in, const int* in_sizes, int n_in,
                              void* d_out, int out_size, void* d_ws, size_t ws_size,
                              hipStream_t stream) {
  const float* obs   = (const float*)d_in[0];
  const float* x0    = (const float*)d_in[1];
  const float* A_T   = (const float*)d_in[2];
  const float* By_T  = (const float*)d_in[3];
  const float* Cu_T  = (const float*)d_in[4];
  const float* Duy_T = (const float*)d_in[5];
  const float* lstd  = (const float*)d_in[6];
  const float* W0    = (const float*)d_in[7];
  const float* b0v   = (const float*)d_in[8];
  const float* W1    = (const float*)d_in[9];
  const float* b1v   = (const float*)d_in[10];
  const float* W2    = (const float*)d_in[11];
  const float* b2v   = (const float*)d_in[12];
  float* outp = (float*)d_out;

  // workspace layout: f32 scratch then f16 packs (all 16B aligned)
  float* bufA   = (float*)d_ws;        // 65536
  float* bufB   = bufA + 65536;        // 65536
  float* bufPhi = bufB + 65536;        // 65536
  float* bufGam = bufPhi + 65536;      // 32768
  _Float16* PhiP = (_Float16*)(bufGam + 32768);  // 65536 h
  _Float16* GamP = PhiP + 65536;                 // 32768 h
  _Float16* CuP  = GamP + 32768;                 // 8192 h
  _Float16* DuyP = CuP + 8192;                   // 4096 h
  _Float16* W0P  = DuyP + 4096;                  // 8192 h
  _Float16* W1P  = W0P + 8192;                   // 4096 h

  const float h = 0.01f;
  // Horner: S1=I+h/4*M; S2=I+h/3*M@S1; S3=I+h/2*M@S2; Phi=I+h*M@S3; Gam=h*By_T@S3
  eltI_f32<<<256, 256, 0, stream>>>(A_T, bufA, h / 4.0f, Nsz);
  mm_f32<<<256, 256, 0, stream>>>(A_T, bufA, bufB, Nsz, Nsz, Nsz, h / 3.0f, 1);
  mm_f32<<<256, 256, 0, stream>>>(A_T, bufB, bufA, Nsz, Nsz, Nsz, h / 2.0f, 1);
  mm_f32<<<256, 256, 0, stream>>>(A_T, bufA, bufPhi, Nsz, Nsz, Nsz, h, 1);
  mm_f32<<<128, 256, 0, stream>>>(By_T, bufA, bufGam, OBSsz, Nsz, Nsz, h, 0);

  pack_b_f16<<<256, 256, 0, stream>>>(bufPhi, PhiP, Nsz, Nsz);
  pack_b_f16<<<128, 256, 0, stream>>>(bufGam, GamP, OBSsz, Nsz);
  pack_b_f16<<<32, 256, 0, stream>>>(Cu_T, CuP, Nsz, OUTsz);
  pack_b_f16<<<16, 256, 0, stream>>>(Duy_T, DuyP, OBSsz, OUTsz);
  pack_b_f16<<<32, 256, 0, stream>>>(W0, W0P, OBSsz, Hsz);
  pack_b_f16<<<16, 256, 0, stream>>>(W1, W1P, Hsz, Hsz);

  // persistent scan: 2 WGs x 512 threads, 240KB dynamic LDS each
  (void)hipFuncSetAttribute(reinterpret_cast<const void*>(scan_kernel),
                            hipFuncAttributeMaxDynamicSharedMemorySize, 245760);
  scan_kernel<<<2, 512, 245760, stream>>>(obs, x0, outp, PhiP);

  fill_logstd<<<(Bsz * Tsz * OUTsz + 255) / 256, 256, 0, stream>>>(lstd, outp);

  (void)hipFuncSetAttribute(reinterpret_cast<const void*>(mlp_kernel),
                            hipFuncAttributeMaxDynamicSharedMemorySize, 90880);
  mlp_kernel<<<256, 512, 90880, stream>>>(obs, W0P, b0v, b1v, W2, b2v,
                                          outp + (size_t)Bsz * Tsz * 64 + Bsz * Nsz);
}
